// ConvolutionalLayer_24283745092168
// MI455X (gfx1250) — compile-verified, matrix operand order checked by reading, not run on hardware
//
#include <hip/hip_runtime.h>

// ---------------------------------------------------------------------------
// y = sum_r A_r @ X @ W_r^T   with  A:[8,4096,4096] f32, X:[4096,128] f32,
//                                   W:[8,128,128] f32  ->  y:[4096,128] f32
//
// Strategy (MI455X, bandwidth-bound at ~23us for the 512MB adjacency stream):
//   1) pack_xw_kernel : Z_r = X @ W_r^T via v_wmma_f32_16x16x32_bf16, stored
//      in d_ws directly in the WMMA *B-fragment* layout (8 MB, stays in L2).
//   2) rgcn_main_kernel: y += A_r @ Z_r ; A is loaded fp32 (each byte exactly
//      once from HBM), converted to bf16 in registers, WMMA bf16 accumulate
//      in f32, results merged across relations with global f32 atomics.
// ---------------------------------------------------------------------------

typedef __attribute__((ext_vector_type(8)))  float  v8f;
typedef __attribute__((ext_vector_type(16))) __bf16 v16bf;

#define RREL  8
#define NN    4096
#define FF    128
#define KSTEP 32
#define NKB   (NN / KSTEP)   // 128 K-blocks per relation
#define NCB   (FF / 16)      // 8 column blocks
// packed Z fragment: [r][kb][cb][lane:32][16 bf16]  = 512 elems (1 KB) each

// Load a 16-lane-group x 32-K fragment of fp32 data and convert to the
// v16bf A/B register layout for v_wmma_f32_16x16x32_bf16.
// Lanes 0-15 hold K = {kbase..kbase+7, kbase+16..kbase+23},
// lanes 16-31 hold K = {kbase+8..+15, kbase+24..+31}   (lhalf = lane>>4).
static __device__ __forceinline__ v16bf load_frag_f32(const float* __restrict__ rowPtr,
                                                      int kbase, int lhalf) {
  v8f g0 = *(const v8f*)(rowPtr + kbase + 8 * lhalf);
  v8f g1 = *(const v8f*)(rowPtr + kbase + 16 + 8 * lhalf);
  v16bf a;
#pragma unroll
  for (int i = 0; i < 8; ++i) {
    a[i]     = (__bf16)g0[i];
    a[i + 8] = (__bf16)g1[i];
  }
  return a;
}

// ---------------------------------------------------------------------------
// Kernel 1: Z[r][m][o] = sum_f X[m][f] * W[r][o][f], written in packed
// B-fragment layout.  One wave per output fragment (r, kb, cb): it computes
// two adjacent 16x16 WMMA C-tiles (m-blocks 2kb, 2kb+1) whose C-register
// layout maps lane-for-lane onto the B-fragment needed by kernel 2.
// 8192 fragments -> 1024 blocks x 8 waves.
// ---------------------------------------------------------------------------
__global__ void __launch_bounds__(256)
pack_xw_kernel(const float* __restrict__ X, const float* __restrict__ W,
               __bf16* __restrict__ Z) {
  const int wid   = threadIdx.x >> 5;
  const int lane  = threadIdx.x & 31;
  const int lrow  = lane & 15;
  const int lhalf = lane >> 4;

  int frag = blockIdx.x * 8 + wid;          // [0, 8192)
  const int cb = frag & 7;   frag >>= 3;
  const int kb = frag & 127; frag >>= 7;
  const int r  = frag;                      // [0, 8)
  const int m0 = kb * KSTEP;
  const int o0 = cb * 16;

  const float* xrow0 = X + (size_t)(m0 + lrow) * FF;        // A of tile0
  const float* xrow1 = X + (size_t)(m0 + 16 + lrow) * FF;   // A of tile1
  const float* wrow  = W + ((size_t)(r * FF) + (o0 + lrow)) * FF; // B (=W^T)

  v8f c0 = {}, c1 = {};
#pragma unroll
  for (int f = 0; f < FF; f += KSTEP) {
    v16bf b  = load_frag_f32(wrow,  f, lhalf);
    v16bf a0 = load_frag_f32(xrow0, f, lhalf);
    v16bf a1 = load_frag_f32(xrow1, f, lhalf);
    c0 = __builtin_amdgcn_wmma_f32_16x16x32_bf16(false, a0, false, b,
                                                 (short)0, c0, false, false);
    c1 = __builtin_amdgcn_wmma_f32_16x16x32_bf16(false, a1, false, b,
                                                 (short)0, c1, false, false);
  }

  // C-fragment -> packed B-fragment (no cross-lane movement needed):
  // lane n (n<16):  e=v   <-> k_local = v        (from c0)
  //                 e=v+8 <-> k_local = 16 + v   (from c1)
  // lane n+16:      e=v   <-> k_local = 8  + v   (from c0)
  //                 e=v+8 <-> k_local = 24 + v   (from c1)
  v16bf outv;
#pragma unroll
  for (int v = 0; v < 8; ++v) {
    outv[v]     = (__bf16)c0[v];
    outv[v + 8] = (__bf16)c1[v];
  }
  __bf16* base = Z + ((size_t)((r * NKB + kb) * NCB + cb) << 9); // *512 elems
  *(v16bf*)(base + lane * 16) = outv;
}

// ---------------------------------------------------------------------------
// Kernel 2: y += A_r @ Z_r.  grid = (4096/128 row tiles, 8 relations),
// block = 256 threads = 8 waves; each wave computes 16 rows x 128 cols
// (8 f32 accumulators).  A streamed fp32 from HBM exactly once; Z-fragments
// are contiguous 32B/lane loads that hit L2/WGP$.
// ---------------------------------------------------------------------------
__global__ void __launch_bounds__(256)
rgcn_main_kernel(const float* __restrict__ A, const __bf16* __restrict__ Z,
                 float* __restrict__ Y) {
  const int wid   = threadIdx.x >> 5;
  const int lane  = threadIdx.x & 31;
  const int lrow  = lane & 15;
  const int lhalf = lane >> 4;
  const int r     = blockIdx.y;
  const int row0  = blockIdx.x * 128 + wid * 16;   // this wave's 16 rows

  const float*  arow = A + ((size_t)r * NN + (row0 + lrow)) * NN;
  const __bf16* zr   = Z + ((size_t)(r * NKB * NCB) << 9) + lane * 16;

  v8f acc[NCB] = {};

  for (int kb = 0; kb < NKB; ++kb) {
    const int k = kb * KSTEP;
    if (kb + 2 < NKB)  // prefetch A two K-steps ahead -> global_prefetch_b8
      __builtin_prefetch(arow + k + 2 * KSTEP, 0, 0);

    v16bf a = load_frag_f32(arow, k, lhalf);
    const __bf16* zkb = zr + ((size_t)(kb * NCB) << 9);
#pragma unroll
    for (int cb = 0; cb < NCB; ++cb) {
      v16bf b = *(const v16bf*)(zkb + ((size_t)cb << 9));
      acc[cb] = __builtin_amdgcn_wmma_f32_16x16x32_bf16(false, a, false, b,
                                                        (short)0, acc[cb],
                                                        false, false);
    }
  }

  // C layout: lane holds col = cb*16 + lrow, rows row0 + 8*lhalf + v.
  float* ybase = Y + (size_t)(row0 + 8 * lhalf) * FF + lrow;
#pragma unroll
  for (int cb = 0; cb < NCB; ++cb) {
#pragma unroll
    for (int v = 0; v < 8; ++v) {
      unsafeAtomicAdd(ybase + (size_t)v * FF + cb * 16, acc[cb][v]);
    }
  }
}

// ---------------------------------------------------------------------------
extern "C" void kernel_launch(void* const* d_in, const int* in_sizes, int n_in,
                              void* d_out, int out_size, void* d_ws, size_t ws_size,
                              hipStream_t stream) {
  const float* A = (const float*)d_in[0];   // adjacency [8,4096,4096]
  const float* X = (const float*)d_in[1];   // features  [4096,128]
  const float* W = (const float*)d_in[2];   // weight    [8,128,128]
  float*  Y = (float*)d_out;                // [4096,128]
  __bf16* Z = (__bf16*)d_ws;                // packed Z fragments: 8 MB

  hipMemsetAsync(d_out, 0, (size_t)out_size * sizeof(float), stream);
  pack_xw_kernel<<<dim3(1024), dim3(256), 0, stream>>>(X, W, Z);
  rgcn_main_kernel<<<dim3(NN / 128, RREL), dim3(256), 0, stream>>>(A, Z, Y);
}
